// AttentionBlock_4604204941407
// MI455X (gfx1250) — matrix operand order repaired and checked by reading, not compiled, not run on hardware
//
#include <hip/hip_runtime.h>
#include <hip/hip_bf16.h>

// ---------------------------------------------------------------------------
// Fused Swin-style window attention block for MI455X (gfx1250, wave32, WMMA).
// One workgroup (256 threads = 8 waves) per 8x8 window (64 tokens x 512 ch).
// Residual tile t kept in LDS fp32; all GEMMs via v_wmma_f32_16x16x32_f16.
// Weights pre-converted fp32->f16 and pre-swizzled into B-fragment layout.
// Gather/scatter and LayerNorm vectorized to b128 global / b64 LDS accesses.
// ---------------------------------------------------------------------------

typedef __attribute__((ext_vector_type(16))) _Float16 v16h;
typedef __attribute__((ext_vector_type(4)))  _Float16 v4h;
typedef __attribute__((ext_vector_type(8)))  float    v8f;

#define C_DIM  512
#define NH     8
#define HD     64
#define L_TOK  64
#define T_LD   520   // f32 LDS stride for residual t (bank spread)
#define X_LD   520   // f16 LDS stride for xn / ob
#define S_LD   68    // f32 LDS stride for scores
#define A_LD   72    // f16 LDS stride for per-head q/k/v/attn

__device__ __forceinline__ int lane_id() { return (int)(threadIdx.x & 31u); }

// ---- WMMA wrapper ----------------------------------------------------------
__device__ __forceinline__ v8f wmma_step(v16h a, v16h b, v8f c) {
  return __builtin_amdgcn_wmma_f32_16x16x32_f16(
      /*neg_a=*/false, a, /*neg_b=*/false, b,
      /*c_mod=*/(short)0, c, /*reuse_a=*/false, /*reuse_b=*/false);
}

// ---- A fragment (16x32 f16) from LDS, row-major with stride lda (halves) ---
// lane L (L<16): row m0+L, K = k0 + {0..15}; lane L>=16: K = k0+{16..31}
__device__ __forceinline__ v16h ldA(const _Float16* A, int lda, int m0, int k0) {
  int L = lane_id();
  int r = m0 + (L & 15);
  int g = (L >> 4) & 1;
  const _Float16* p0 = A + r * lda + k0 + g * 16;
  const _Float16* p1 = p0 + 8;
  v16h a;
#pragma unroll
  for (int e = 0; e < 8; ++e) a[e] = p0[e];
#pragma unroll
  for (int e = 0; e < 8; ++e) a[8 + e] = p1[e];
  return a;
}

// ---- B fragment (32x16 f16) from row-major K x N buffer (stride ldb) -------
__device__ __forceinline__ v16h ldB(const _Float16* B, int ldb, int k0, int n0) {
  int L = lane_id();
  int n = n0 + (L & 15);
  int kk = k0 + ((L >> 4) & 1) * 16;
  v16h b;
#pragma unroll
  for (int e = 0; e < 16; ++e) b[e] = B[(kk + e) * ldb + n];
  return b;
}

// ---- B fragment where logical B[k][n] = Bm[n][k] (transposed source) -------
__device__ __forceinline__ v16h ldBT(const _Float16* Bm, int ldb, int k0, int n0) {
  int L = lane_id();
  int n = n0 + (L & 15);
  int kk = k0 + ((L >> 4) & 1) * 16;
  const _Float16* p = Bm + n * ldb + kk;
  v16h b;
#pragma unroll
  for (int e = 0; e < 16; ++e) b[e] = p[e];
  return b;
}

// ---- B fragment from pre-swizzled f16 weights ------------------------------
// layout: tile (kt = k/32, nt = n/16): [(kt*ntiles+nt)*32 + lane][16 halves]
__device__ __forceinline__ v16h ldBsw(const _Float16* W, int ntiles, int kt, int nt) {
  const _Float16* p = W + (((size_t)(kt * ntiles + nt)) * 32 + (size_t)lane_id()) * 16;
  return *(const v16h*)p;
}

// ---- C/D fragment stores/loads (16x16, lane L: N=L&15, M=(L>>4)*8+r) -------
__device__ __forceinline__ void stC_f(float* D, int ldd, int m0, int n0, v8f c) {
  int L = lane_id();
  int n = n0 + (L & 15);
  int mb = m0 + ((L >> 4) & 1) * 8;
#pragma unroll
  for (int r = 0; r < 8; ++r) D[(mb + r) * ldd + n] = c[r];
}
__device__ __forceinline__ void stC_h(_Float16* D, int ldd, int m0, int n0, v8f c) {
  int L = lane_id();
  int n = n0 + (L & 15);
  int mb = m0 + ((L >> 4) & 1) * 8;
#pragma unroll
  for (int r = 0; r < 8; ++r) D[(mb + r) * ldd + n] = (_Float16)c[r];
}
__device__ __forceinline__ v8f ldC_f(const float* D, int ldd, int m0, int n0) {
  int L = lane_id();
  int n = n0 + (L & 15);
  int mb = m0 + ((L >> 4) & 1) * 8;
  v8f c;
#pragma unroll
  for (int r = 0; r < 8; ++r) c[r] = D[(mb + r) * ldd + n];
  return c;
}
__device__ __forceinline__ void addC_f(float* D, int ldd, int m0, int n0, v8f c) {
  int L = lane_id();
  int n = n0 + (L & 15);
  int mb = m0 + ((L >> 4) & 1) * 8;
#pragma unroll
  for (int r = 0; r < 8; ++r) D[(mb + r) * ldd + n] += c[r];
}

// ---- per-token LayerNorm over C=512, fp32 LDS in, f16 LDS out --------------
// pass 1: 4 threads/row, float4 reads; pass 2: each thread owns a contiguous
// 128-channel run of one row -> float4 LDS/global reads, v4h (b64) LDS writes.
__device__ __forceinline__ void layer_norm(const float* tb, _Float16* dst,
                                           const float* __restrict__ gamma,
                                           const float* __restrict__ beta,
                                           float (*red)[8], float (*murs)[2]) {
  int row = (int)(threadIdx.x >> 2);
  int sub = (int)(threadIdx.x & 3);
  const float* tr = tb + row * T_LD + sub * 128;
  float s = 0.f, ss = 0.f;
#pragma unroll 4
  for (int c = 0; c < 128; c += 4) {
    float4 v = *(const float4*)(tr + c);
    s  += v.x + v.y + v.z + v.w;
    ss += v.x * v.x + v.y * v.y + v.z * v.z + v.w * v.w;
  }
  red[row][sub] = s;
  red[row][4 + sub] = ss;
  __syncthreads();
  if (sub == 0) {
    float S = red[row][0] + red[row][1] + red[row][2] + red[row][3];
    float Q = red[row][4] + red[row][5] + red[row][6] + red[row][7];
    float mu = S * (1.f / 512.f);
    float var = Q * (1.f / 512.f) - mu * mu;
    murs[row][0] = mu;
    murs[row][1] = rsqrtf(var + 1e-5f);
  }
  __syncthreads();
  {
    int l  = (int)(threadIdx.x & 63u);
    int c0 = ((int)threadIdx.x >> 6) * 128;
    float mu = murs[l][0], rs = murs[l][1];
    const float* src = tb + l * T_LD + c0;
    _Float16* d = dst + l * X_LD + c0;
#pragma unroll 4
    for (int c = 0; c < 128; c += 4) {
      float4 v = *(const float4*)(src + c);
      float4 g = *(const float4*)(gamma + c0 + c);
      float4 bt = *(const float4*)(beta + c0 + c);
      v4h o;
      o[0] = (_Float16)((v.x - mu) * rs * g.x + bt.x);
      o[1] = (_Float16)((v.y - mu) * rs * g.y + bt.y);
      o[2] = (_Float16)((v.z - mu) * rs * g.z + bt.z);
      o[3] = (_Float16)((v.w - mu) * rs * g.w + bt.w);
      *(v4h*)(d + c) = o;
    }
  }
  __syncthreads();
}

// ---------------------------------------------------------------------------
// Weight conversion + swizzle kernel: fp32 row-major (K x N) -> f16 fragments
// ---------------------------------------------------------------------------
__global__ void pack_w(const float* __restrict__ src, _Float16* __restrict__ dst,
                       int K, int N) {
  int i = blockIdx.x * 256 + threadIdx.x;
  if (i >= K * N) return;
  int ntiles = N >> 4;
  int tile = i >> 9;          // 512 elements per 32x16 tile
  int lane = (i >> 4) & 31;
  int e = i & 15;
  int kt = tile / ntiles, nt = tile % ntiles;
  int k = kt * 32 + ((lane >> 4) & 1) * 16 + e;
  int n = nt * 16 + (lane & 15);
  dst[i] = (_Float16)src[(size_t)k * N + n];
}

// ---------------------------------------------------------------------------
// Main fused kernel: one workgroup per window
// ---------------------------------------------------------------------------
__global__ __launch_bounds__(256)
void win_attn_block(const float* __restrict__ x,
                    const float* __restrict__ g1, const float* __restrict__ be1,
                    const float* __restrict__ g2, const float* __restrict__ be2,
                    const float* __restrict__ b_qkv, const float* __restrict__ b_out,
                    const float* __restrict__ b1, const float* __restrict__ b2,
                    const _Float16* __restrict__ wqkv, const _Float16* __restrict__ wout,
                    const _Float16* __restrict__ w1, const _Float16* __restrict__ w2,
                    float* __restrict__ y) {
  __shared__ float    t_s[L_TOK * T_LD];   // residual, fp32 (133,120 B)
  __shared__ _Float16 xn[L_TOK * X_LD];    // LN output / MLP input (66,560 B)
  __shared__ _Float16 ob[L_TOK * X_LD];    // attn-out, then MLP hidden (66,560 B)
  __shared__ _Float16 qb[L_TOK * A_LD];    // per-head q (9,216 B)
  __shared__ _Float16 kb[L_TOK * A_LD];    // per-head k, then attn weights
  __shared__ _Float16 vb[L_TOK * A_LD];    // per-head v
  __shared__ float    sc[L_TOK * S_LD];    // scores fp32 (17,408 B)
  __shared__ float    red[L_TOK][8];
  __shared__ float    murs[L_TOK][2];

  const int widx = blockIdx.x;
  const int b  = widx >> 8;      // 256 windows per batch image
  const int n  = widx & 255;
  const int wy = n >> 4, wx = n & 15;
  const int wave = (int)(threadIdx.x >> 5);
  const int tid  = (int)threadIdx.x;

  // ---- gather window: x(B,C,128,128) -> t_s[l][c], b128 loads --------------
  // 4096 (c,iy) row-pairs of 8 contiguous floats; 2 float4 per pair.
  const float* xb = x + (((size_t)b * C_DIM) * 128 + (size_t)(wy * 8)) * 128 + wx * 8;
#pragma unroll 4
  for (int i = 0; i < 32; ++i) {
    int idx  = i * 256 + tid;        // 0..8191
    int half = idx & 1;
    int pair = idx >> 1;             // 0..4095
    int iy = pair & 7, c = pair >> 3;
    float4 v = *(const float4*)(xb + (size_t)c * 16384 + iy * 128 + half * 4);
    int l = iy * 8 + half * 4;
    t_s[(l + 0) * T_LD + c] = v.x;
    t_s[(l + 1) * T_LD + c] = v.y;
    t_s[(l + 2) * T_LD + c] = v.z;
    t_s[(l + 3) * T_LD + c] = v.w;
  }
  __syncthreads();

  // ---- LN1 -> xn (f16) -----------------------------------------------------
  layer_norm(t_s, xn, g1, be1, red, murs);

  // ---- attention, one head at a time --------------------------------------
  for (int h = 0; h < NH; ++h) {
    // QKV projection for this head: 12 N-tiles (4 q, 4 k, 4 v), K=512
    for (int nt = wave; nt < 12; nt += 8) {
      int part = nt >> 2;                 // 0=q 1=k 2=v
      int sub  = nt & 3;
      int wcol = part * 512 + h * 64 + sub * 16;
      _Float16* dst = (part == 0) ? qb : (part == 1) ? kb : vb;
      float bias = b_qkv[wcol + (lane_id() & 15)];
      v8f c0, c1, c2, c3;
#pragma unroll
      for (int r = 0; r < 8; ++r) { c0[r] = bias; c1[r] = bias; c2[r] = bias; c3[r] = bias; }
      int wnt = wcol >> 4;
      for (int kk = 0; kk < 16; ++kk) {
        v16h bf = ldBsw(wqkv, 96, kk, wnt);
        c0 = wmma_step(ldA(xn, X_LD,  0, kk * 32), bf, c0);
        c1 = wmma_step(ldA(xn, X_LD, 16, kk * 32), bf, c1);
        c2 = wmma_step(ldA(xn, X_LD, 32, kk * 32), bf, c2);
        c3 = wmma_step(ldA(xn, X_LD, 48, kk * 32), bf, c3);
      }
      stC_h(dst, A_LD,  0, sub * 16, c0);
      stC_h(dst, A_LD, 16, sub * 16, c1);
      stC_h(dst, A_LD, 32, sub * 16, c2);
      stC_h(dst, A_LD, 48, sub * 16, c3);
    }
    __syncthreads();

    // scores = q @ k^T / (64*64)  -> sc (fp32)
    for (int tt = wave; tt < 16; tt += 8) {
      int mt = tt >> 2, nt = tt & 3;
      v8f c;
#pragma unroll
      for (int r = 0; r < 8; ++r) c[r] = 0.f;
      for (int kk = 0; kk < 2; ++kk)
        c = wmma_step(ldA(qb, A_LD, mt * 16, kk * 32),
                      ldBT(kb, A_LD, kk * 32, nt * 16), c);
#pragma unroll
      for (int r = 0; r < 8; ++r) c[r] *= (1.0f / 4096.0f);
      stC_f(sc, S_LD, mt * 16, nt * 16, c);
    }
    __syncthreads();

    // row softmax -> attn weights (f16) written into kb (k no longer needed)
    if (tid < 64) {
      float* srow = sc + tid * S_LD;
      float m = srow[0];
      for (int j = 1; j < 64; ++j) m = fmaxf(m, srow[j]);
      float s = 0.f;
      for (int j = 0; j < 64; ++j) { float e = __expf(srow[j] - m); srow[j] = e; s += e; }
      float inv = 1.0f / s;
      _Float16* arow = kb + tid * A_LD;
      for (int j = 0; j < 64; ++j) arow[j] = (_Float16)(srow[j] * inv);
    }
    __syncthreads();

    // o_h = attn @ v  -> ob[:, h*64 ...]
    for (int tt = wave; tt < 16; tt += 8) {
      int mt = tt >> 2, nt = tt & 3;
      v8f c;
#pragma unroll
      for (int r = 0; r < 8; ++r) c[r] = 0.f;
      for (int kk = 0; kk < 2; ++kk)
        c = wmma_step(ldA(kb, A_LD, mt * 16, kk * 32),
                      ldB(vb, A_LD, kk * 32, nt * 16), c);
      stC_h(ob, X_LD, mt * 16, h * 64 + nt * 16, c);
    }
    __syncthreads();
  }

  // ---- out projection: t += ob @ w_out + b_out -----------------------------
  for (int tt = wave; tt < 128; tt += 8) {
    int mt = tt >> 5, nt = tt & 31;
    float bias = b_out[nt * 16 + (lane_id() & 15)];
    v8f c;
#pragma unroll
    for (int r = 0; r < 8; ++r) c[r] = bias;
    for (int kk = 0; kk < 16; ++kk)
      c = wmma_step(ldA(ob, X_LD, mt * 16, kk * 32), ldBsw(wout, 32, kk, nt), c);
    addC_f(t_s, T_LD, mt * 16, nt * 16, c);
  }
  __syncthreads();

  // ---- LN2 -> xn (f16) -----------------------------------------------------
  layer_norm(t_s, xn, g2, be2, red, murs);

  // ---- MLP in 4 hidden chunks of 512 --------------------------------------
  for (int cc = 0; cc < 4; ++cc) {
    // h = gelu(xn @ w1[:, cc*512 : cc*512+512] + b1)  -> ob (f16)
    for (int tt = wave; tt < 128; tt += 8) {
      int mt = tt >> 5, nt = tt & 31;
      float bias = b1[cc * 512 + nt * 16 + (lane_id() & 15)];
      v8f c;
#pragma unroll
      for (int r = 0; r < 8; ++r) c[r] = bias;
      for (int kk = 0; kk < 16; ++kk)
        c = wmma_step(ldA(xn, X_LD, mt * 16, kk * 32),
                      ldBsw(w1, 128, kk, cc * 32 + nt), c);
#pragma unroll
      for (int r = 0; r < 8; ++r) {
        float v = c[r];
        c[r] = 0.5f * v * (1.0f + erff(v * 0.70710678118f));  // exact GELU
      }
      stC_h(ob, X_LD, mt * 16, nt * 16, c);
    }
    __syncthreads();

    // t += h @ w2[cc*512 : , :]  (+ b2 once)
    for (int tt = wave; tt < 128; tt += 8) {
      int mt = tt >> 5, nt = tt & 31;
      v8f c = ldC_f(t_s, T_LD, mt * 16, nt * 16);
      if (cc == 0) {
        float bias = b2[nt * 16 + (lane_id() & 15)];
#pragma unroll
        for (int r = 0; r < 8; ++r) c[r] += bias;
      }
      for (int kk = 0; kk < 16; ++kk)
        c = wmma_step(ldA(ob, X_LD, mt * 16, kk * 32),
                      ldBsw(w2, 32, cc * 16 + kk, nt), c);
      stC_f(t_s, T_LD, mt * 16, nt * 16, c);
    }
    __syncthreads();
  }

  // ---- scatter t -> y (window unpartition), b128 stores --------------------
  float* yb = y + (((size_t)b * C_DIM) * 128 + (size_t)(wy * 8)) * 128 + wx * 8;
#pragma unroll 4
  for (int i = 0; i < 32; ++i) {
    int idx  = i * 256 + tid;
    int half = idx & 1;
    int pair = idx >> 1;
    int iy = pair & 7, c = pair >> 3;
    int l = iy * 8 + half * 4;
    float4 v;
    v.x = t_s[(l + 0) * T_LD + c];
    v.y = t_s[(l + 1) * T_LD + c];
    v.z = t_s[(l + 2) * T_LD + c];
    v.w = t_s[(l + 3) * T_LD + c];
    *(float4*)(yb + (size_t)c * 16384 + iy * 128 + half * 4) = v;
  }
}

// ---------------------------------------------------------------------------
// Launch
// ---------------------------------------------------------------------------
extern "C" void kernel_launch(void* const* d_in, const int* in_sizes, int n_in,
                              void* d_out, int out_size, void* d_ws, size_t ws_size,
                              hipStream_t stream) {
  (void)in_sizes; (void)n_in; (void)out_size; (void)ws_size;
  const float* x     = (const float*)d_in[0];
  const float* g1    = (const float*)d_in[1];
  const float* be1   = (const float*)d_in[2];
  const float* g2    = (const float*)d_in[3];
  const float* be2   = (const float*)d_in[4];
  const float* w_qkv = (const float*)d_in[5];
  const float* b_qkv = (const float*)d_in[6];
  const float* w_out = (const float*)d_in[7];
  const float* b_out = (const float*)d_in[8];
  const float* w1    = (const float*)d_in[9];
  const float* b1    = (const float*)d_in[10];
  const float* w2    = (const float*)d_in[11];
  const float* b2    = (const float*)d_in[12];

  _Float16* ws16    = (_Float16*)d_ws;
  _Float16* wqkv_h  = ws16;                 // 512*1536 = 786432 halves
  _Float16* wout_h  = ws16 + 786432;        // 512*512  = 262144
  _Float16* w1_h    = ws16 + 1048576;       // 512*2048 = 1048576
  _Float16* w2_h    = ws16 + 2097152;       // 2048*512 = 1048576  (total 6 MB)

  pack_w<<<3072, 256, 0, stream>>>(w_qkv, wqkv_h, 512, 1536);
  pack_w<<<1024, 256, 0, stream>>>(w_out, wout_h, 512, 512);
  pack_w<<<4096, 256, 0, stream>>>(w1,    w1_h,   512, 2048);
  pack_w<<<4096, 256, 0, stream>>>(w2,    w2_h,   2048, 512);

  win_attn_block<<<2048, 256, 0, stream>>>(
      x, g1, be1, g2, be2, b_qkv, b_out, b1, b2,
      wqkv_h, wout_h, w1_h, w2_h, (float*)d_out);
}
